// GPT2Attention_8993661518663
// MI455X (gfx1250) — compile-verified, matrix operand order checked by reading, not run on hardware
//
#include <hip/hip_runtime.h>

#define SEQ 2048
#define HID 1024
#define NH 16
#define HD 64

typedef __attribute__((ext_vector_type(16))) __bf16 bf16x16;
typedef __attribute__((ext_vector_type(8)))  __bf16 bf16x8;
typedef __attribute__((ext_vector_type(8)))  float  f32x8;

// D = A(16x32 bf16) * B(32x16 bf16) + C(16x16 f32), wave32
__device__ inline f32x8 wmma_bf16(bf16x16 a, bf16x16 b, f32x8 c) {
  return __builtin_amdgcn_wmma_f32_16x16x32_bf16(
      /*neg_a=*/false, a, /*neg_b=*/false, b,
      /*c_mod=*/(short)0, c, /*reuse_a=*/false, /*reuse_b=*/false);
}

// load 8 contiguous bf16 into fragment slots [base, base+8)
__device__ inline void load8_bf(bf16x16 &d, int base, const __bf16 *p) {
  bf16x8 v = *(const bf16x8 *)p;
#pragma unroll
  for (int i = 0; i < 8; ++i) d[base + i] = v[i];
}

// load 8 contiguous f32, convert to bf16 into fragment slots [base, base+8)
__device__ inline void cvt8_f(bf16x16 &d, int base, const float *p) {
  float4 a = ((const float4 *)p)[0];
  float4 b = ((const float4 *)p)[1];
  d[base + 0] = (__bf16)a.x; d[base + 1] = (__bf16)a.y;
  d[base + 2] = (__bf16)a.z; d[base + 3] = (__bf16)a.w;
  d[base + 4] = (__bf16)b.x; d[base + 5] = (__bf16)b.y;
  d[base + 6] = (__bf16)b.z; d[base + 7] = (__bf16)b.w;
}

// cross-lane xor shuffle within groups of 32 (and_mask=0x1f, xor<16 keeps halves)
#define SWIZ_XOR_F(v, mask) \
  __int_as_float(__builtin_amdgcn_ds_swizzle(__float_as_int(v), (((mask) << 10) | 0x1f)))

// WT[n][k] = (bf16) W[k][n], 32x32 LDS-tiled so reads & writes are coalesced.
__global__ __launch_bounds__(256) void transpose_cvt_kernel(
    const float *__restrict__ W, __bf16 *__restrict__ WT, int K, int N) {
  __shared__ __bf16 tile[32][33];
  const int bk = blockIdx.x * 32;
  const int bn = blockIdx.y * 32;
  const int tx = threadIdx.x & 31;
  const int ty = threadIdx.x >> 5;  // 0..7
#pragma unroll
  for (int i = 0; i < 4; ++i) {
    int k = bk + ty + 8 * i;
    tile[ty + 8 * i][tx] = (__bf16)W[(size_t)k * N + bn + tx];
  }
  __syncthreads();
#pragma unroll
  for (int i = 0; i < 4; ++i) {
    int n = bn + ty + 8 * i;
    WT[(size_t)n * K + bk + tx] = tile[tx][ty + 8 * i];
  }
}

// C[M,N] = A[M,K](f32) * BT[N,K](bf16)^T + bias ; 4 waves, 64x64 tile/block.
// Double-buffered K loop (K multiple of 64, K>=128).
// mode 0: store f32 C row-major. mode 1: scatter QKV into Q/K/VT bf16 buffers.
__global__ __launch_bounds__(128) void gemm_wmma_kernel(
    const float *__restrict__ A, const __bf16 *__restrict__ BT,
    const float *__restrict__ bias, int M, int N, int K, int mode,
    float *__restrict__ C, __bf16 *__restrict__ Qo, __bf16 *__restrict__ Ko,
    __bf16 *__restrict__ VTo) {
  const int wave = threadIdx.x >> 5;
  const int lane = threadIdx.x & 31;
  const int hi = lane >> 4;
  const int ln = lane & 15;
  const int kg0 = hi * 8;                 // per-lane K sub-offset
  const int bm = blockIdx.y * 64;
  const int bn = blockIdx.x * 64;

  f32x8 acc[4];
#pragma unroll
  for (int t = 0; t < 4; ++t)
#pragma unroll
    for (int i = 0; i < 8; ++i) acc[t][i] = 0.0f;

  const float *arow = A + (size_t)(bm + wave * 16 + ln) * K;
  const __bf16 *brow[4];
#pragma unroll
  for (int t = 0; t < 4; ++t) brow[t] = BT + (size_t)(bn + t * 16 + ln) * K;

  auto loadA = [&](bf16x16 &af, int kk) {
    cvt8_f(af, 0, arow + kk + kg0);
    cvt8_f(af, 8, arow + kk + kg0 + 16);
  };
  auto loadB = [&](bf16x16 *bf, int kk) {
#pragma unroll
    for (int t = 0; t < 4; ++t) {
      load8_bf(bf[t], 0, brow[t] + kk + kg0);
      load8_bf(bf[t], 8, brow[t] + kk + kg0 + 16);
    }
  };
  auto mma4 = [&](bf16x16 &af, bf16x16 *bf) {
#pragma unroll
    for (int t = 0; t < 4; ++t) acc[t] = wmma_bf16(af, bf[t], acc[t]);
  };

  bf16x16 afA, afB, bfA[4], bfB[4];
  loadA(afA, 0);
  loadB(bfA, 0);
  int kk = 0;
  for (; kk < K - 64; kk += 64) {
    loadA(afB, kk + 32);          // prefetch next 32-K slab
    loadB(bfB, kk + 32);
    mma4(afA, bfA);               // compute current while next is in flight
    loadA(afA, kk + 64);
    loadB(bfA, kk + 64);
    mma4(afB, bfB);
  }
  loadA(afB, K - 32);
  loadB(bfB, K - 32);
  mma4(afA, bfA);
  mma4(afB, bfB);

#pragma unroll
  for (int t = 0; t < 4; ++t) {
    int n = bn + t * 16 + ln;
    float bv = bias[n];
#pragma unroll
    for (int r = 0; r < 8; ++r) {
      int row = bm + wave * 16 + r + hi * 8;
      float val = acc[t][r] + bv;
      if (mode == 0) {
        C[(size_t)row * N + n] = val;
      } else {
        int which = n >> 10;          // 0=Q 1=K 2=V
        int cc = n & 1023;
        int h = cc >> 6;
        int d = cc & 63;
        if (which == 0)
          Qo[(size_t)(h * SEQ + row) * HD + d] = (__bf16)val;
        else if (which == 1)
          Ko[(size_t)(h * SEQ + row) * HD + d] = (__bf16)val;
        else
          VTo[(size_t)(h * HD + d) * SEQ + row] = (__bf16)val;
      }
    }
  }
}

// Flash attention: one wave per (head, 16-query block), 32 keys / iteration.
__global__ __launch_bounds__(32) void attn_kernel(
    const __bf16 *__restrict__ Q, const __bf16 *__restrict__ Kh,
    const __bf16 *__restrict__ VT, float *__restrict__ O) {
  __shared__ __align__(16) __bf16 Plds[16 * 32];
  const int h = blockIdx.y;
  const int qb = blockIdx.x;
  const int lane = threadIdx.x;
  const int hi = lane >> 4;
  const int ln = lane & 15;
  const int kg0 = hi * 8;
  const int q_base = qb * 16;
  const float sm_scale = 0.125f;  // 1/sqrt(64)

  const __bf16 *qrow = Q + (size_t)(h * SEQ + q_base + ln) * HD;
  bf16x16 qa0, qa1;                 // head-dims 0..31 / 32..63
  load8_bf(qa0, 0, qrow + kg0);
  load8_bf(qa0, 8, qrow + kg0 + 16);
  load8_bf(qa1, 0, qrow + 32 + kg0);
  load8_bf(qa1, 8, qrow + 32 + kg0 + 16);

  float mi[8], li[8];
  f32x8 acc[4];
#pragma unroll
  for (int t = 0; t < 4; ++t)
#pragma unroll
    for (int i = 0; i < 8; ++i) acc[t][i] = 0.0f;
#pragma unroll
  for (int r = 0; r < 8; ++r) { mi[r] = -__builtin_inff(); li[r] = 0.0f; }

  const int kv_end = q_base + 16;
  for (int kv = 0; kv < kv_end; kv += 32) {
    // ---- issue all K-fragment loads (one clause), then all V loads ----
    const __bf16 *kr0 = Kh + (size_t)(h * SEQ + kv + ln) * HD;
    const __bf16 *kr1 = Kh + (size_t)(h * SEQ + kv + 16 + ln) * HD;
    bf16x16 kb00, kb01, kb10, kb11;
    load8_bf(kb00, 0, kr0 + kg0);      load8_bf(kb00, 8, kr0 + kg0 + 16);
    load8_bf(kb01, 0, kr0 + 32 + kg0); load8_bf(kb01, 8, kr0 + 32 + kg0 + 16);
    load8_bf(kb10, 0, kr1 + kg0);      load8_bf(kb10, 8, kr1 + kg0 + 16);
    load8_bf(kb11, 0, kr1 + 32 + kg0); load8_bf(kb11, 8, kr1 + 32 + kg0 + 16);

    bf16x16 vf[4];  // independent of scores: overlaps score WMMAs + softmax
#pragma unroll
    for (int t = 0; t < 4; ++t) {
      const __bf16 *vr = VT + (size_t)(h * HD + t * 16 + ln) * SEQ + kv;
      load8_bf(vf[t], 0, vr + kg0);
      load8_bf(vf[t], 8, vr + kg0 + 16);
    }

    f32x8 s0, s1;
#pragma unroll
    for (int i = 0; i < 8; ++i) { s0[i] = 0.0f; s1[i] = 0.0f; }
    s0 = wmma_bf16(qa0, kb00, s0);
    s0 = wmma_bf16(qa1, kb01, s0);
    s1 = wmma_bf16(qa0, kb10, s1);
    s1 = wmma_bf16(qa1, kb11, s1);

    const int col0 = kv + ln, col1 = kv + 16 + ln;
    float p0[8], p1[8];
#pragma unroll
    for (int r = 0; r < 8; ++r) {
      int row = q_base + r + hi * 8;
      float v0 = s0[r] * sm_scale;
      float v1 = s1[r] * sm_scale;
      if (col0 > row) v0 = -__builtin_inff();   // causal mask (cndmask, no EXEC change)
      if (col1 > row) v1 = -__builtin_inff();
      float rm = fmaxf(v0, v1);
      rm = fmaxf(rm, SWIZ_XOR_F(rm, 1));
      rm = fmaxf(rm, SWIZ_XOR_F(rm, 2));
      rm = fmaxf(rm, SWIZ_XOR_F(rm, 4));
      rm = fmaxf(rm, SWIZ_XOR_F(rm, 8));
      float mnew = fmaxf(mi[r], rm);
      float alpha = __expf(mi[r] - mnew);
      mi[r] = mnew;
      p0[r] = __expf(v0 - mnew);
      p1[r] = __expf(v1 - mnew);
      float rs = p0[r] + p1[r];
      rs += SWIZ_XOR_F(rs, 1);
      rs += SWIZ_XOR_F(rs, 2);
      rs += SWIZ_XOR_F(rs, 4);
      rs += SWIZ_XOR_F(rs, 8);
      li[r] = li[r] * alpha + rs;
#pragma unroll
      for (int t = 0; t < 4; ++t) acc[t][r] *= alpha;
    }
    // relayout P: C-fragment (m=r+8hi, n=ln) -> LDS row-major 16x32 bf16
#pragma unroll
    for (int r = 0; r < 8; ++r) {
      Plds[(r + hi * 8) * 32 + ln] = (__bf16)p0[r];
      Plds[(r + hi * 8) * 32 + 16 + ln] = (__bf16)p1[r];
    }
    __syncthreads();
    bf16x16 pf;  // A-fragment of P (16x32)
    load8_bf(pf, 0, &Plds[ln * 32 + kg0]);
    load8_bf(pf, 8, &Plds[ln * 32 + kg0 + 16]);
#pragma unroll
    for (int t = 0; t < 4; ++t) acc[t] = wmma_bf16(pf, vf[t], acc[t]);
    __syncthreads();
  }

#pragma unroll
  for (int r = 0; r < 8; ++r) {
    float inv = 1.0f / li[r];
    int row = q_base + r + hi * 8;
#pragma unroll
    for (int t = 0; t < 4; ++t)
      O[(size_t)row * HID + h * HD + t * 16 + ln] = acc[t][r] * inv;
  }
}

extern "C" void kernel_launch(void *const *d_in, const int *in_sizes, int n_in,
                              void *d_out, int out_size, void *d_ws, size_t ws_size,
                              hipStream_t stream) {
  const float *hidden = (const float *)d_in[0];
  const float *w_attn = (const float *)d_in[1];
  const float *b_attn = (const float *)d_in[2];
  const float *w_proj = (const float *)d_in[3];
  const float *b_proj = (const float *)d_in[4];
  float *out = (float *)d_out;

  char *ws = (char *)d_ws;
  __bf16 *Q    = (__bf16 *)(ws + (size_t)0);            // 4 MiB  [NH][S][HD] bf16
  __bf16 *K    = (__bf16 *)(ws + ((size_t)4  << 20));   // 4 MiB  [NH][S][HD] bf16
  __bf16 *VT   = (__bf16 *)(ws + ((size_t)8  << 20));   // 4 MiB  [NH][HD][S] bf16
  float  *attn = (float  *)(ws + ((size_t)12 << 20));   // 8 MiB  [S][HID] f32
  __bf16 *WTa  = (__bf16 *)(ws + ((size_t)20 << 20));   // 6 MiB  [3H][H] bf16
  __bf16 *WTp  = (__bf16 *)(ws + ((size_t)26 << 20));   // 2 MiB  [H][H]  bf16

  // coalesced tiled transpose+convert of both weight matrices
  transpose_cvt_kernel<<<dim3(HID / 32, 3 * HID / 32), 256, 0, stream>>>(
      w_attn, WTa, HID, 3 * HID);
  transpose_cvt_kernel<<<dim3(HID / 32, HID / 32), 256, 0, stream>>>(
      w_proj, WTp, HID, HID);
  // QKV projection + bias, scatter into Q/K/VT (bf16)
  gemm_wmma_kernel<<<dim3(3 * HID / 64, SEQ / 64), 128, 0, stream>>>(
      hidden, WTa, b_attn, SEQ, 3 * HID, HID, /*mode=*/1, nullptr, Q, K, VT);
  // causal flash attention per (head, 16-row query block)
  attn_kernel<<<dim3(SEQ / 16, NH), 32, 0, stream>>>(Q, K, VT, attn);
  // output projection + bias -> d_out (f32)
  gemm_wmma_kernel<<<dim3(HID / 64, SEQ / 64), 128, 0, stream>>>(
      attn, WTp, b_proj, SEQ, HID, HID, /*mode=*/0, out, nullptr, nullptr, nullptr);
}
